// GraphDecoder_63608465654138
// MI455X (gfx1250) — matrix operand order repaired and checked by reading, not compiled, not run on hardware
//
#include <hip/hip_runtime.h>
#include <hip/hip_bf16.h>
#include <stdint.h>

typedef __attribute__((ext_vector_type(16))) __bf16 v16bf;
typedef __attribute__((ext_vector_type(8)))  __bf16 v8bf;
typedef __attribute__((ext_vector_type(4)))  __bf16 v4bf;
typedef __attribute__((ext_vector_type(8)))  float  v8f;
typedef int v4i __attribute__((vector_size(16)));   // matches builtin param type

#define NB  32            // batch
#define NN  64            // nodes
#define NF  128           // features per node
#define NK  4             // edge types
#define NH1 256           // msg hidden
#define NM  256           // msg out
#define NNH 256           // out-mlp hidden
#define DEG (NN - 1)      // 63 incoming edges per receiver
#define NE  (NN * DEG)    // 4032 edges
#define F2  (2 * NF)      // 256

// ---------------------------------------------------------------------------
// CDNA5 async global->LDS copy (ASYNCcnt path), feature-gated so the kernel
// always compiles; falls back to VGPR-bounce copy if the builtin is absent.
// Builtin prototype (from compiler diagnostic): (v4i*, v4i*, imm int, imm int)
// with generic pointers; order follows global_load_lds precedent:
// (global_src, lds_dst, offset, cpol).
// ---------------------------------------------------------------------------
#if defined(__HIP_DEVICE_COMPILE__) && __has_builtin(__builtin_amdgcn_global_load_async_to_lds_b128)
#define USE_ASYNC_LDS 1
#else
#define USE_ASYNC_LDS 0
#endif

__device__ __forceinline__ void async_b128(const uint4* gsrc, uint4* ldst) {
#if USE_ASYNC_LDS
  __builtin_amdgcn_global_load_async_to_lds_b128((v4i*)gsrc, (v4i*)ldst, 0, 0);
#else
  *ldst = *gsrc;
#endif
}

__device__ __forceinline__ void wait_async_all() {
#if USE_ASYNC_LDS
#if __has_builtin(__builtin_amdgcn_s_wait_asynccnt)
  __builtin_amdgcn_s_wait_asynccnt(0);
#else
  asm volatile("s_wait_asynccnt 0x0" ::: "memory");
#endif
#endif
}

// stage one 64 KB half-panel (4096 x b128) with 256 threads
__device__ __forceinline__ void stage_half(const uint4* __restrict__ gsrc,
                                           uint4* ldst, int tid) {
#pragma unroll 4
  for (int idx = tid; idx < 4096; idx += 256) async_b128(gsrc + idx, ldst + idx);
}

// ---------------------------------------------------------------------------
// Fragment helpers (wave32 layouts per CDNA5 ISA 7.12.2)
// ---------------------------------------------------------------------------
__device__ __forceinline__ v16bf lda(const __bf16* base, int ld, int kbase, int lane) {
  const int h = (lane >> 4) & 1, m = lane & 15;
  const __bf16* p = base + m * ld + kbase + h * 8;
  v8bf lo = *(const v8bf*)p;
  v8bf hi = *(const v8bf*)(p + 16);
  return __builtin_shufflevector(lo, hi, 0, 1, 2, 3, 4, 5, 6, 7,
                                 8, 9, 10, 11, 12, 13, 14, 15);
}

// packed fragment-major (nt-major tile order): one 32B contiguous load per lane
__device__ __forceinline__ v16bf ldbp(const __bf16* Wp, int KT, int kt, int nt, int lane) {
  const __bf16* p = Wp + (((size_t)(nt * KT + kt)) * 32 + lane) * 16;
  return *(const v16bf*)p;
}

// ---------------------------------------------------------------------------
// fp32 -> bf16 weight packing into WMMA B-fragment layout, nt-major tiles.
// ---------------------------------------------------------------------------
__global__ void pack_b_kernel(const float* __restrict__ src, __bf16* __restrict__ dst,
                              int Kdim, int Ndim, int total) {
  int o = blockIdx.x * blockDim.x + threadIdx.x;
  if (o >= total) return;
  const int mat = Kdim * Ndim;
  const int kk = o / mat;
  const int r  = o - kk * mat;
  const int i    = r & 15;
  const int lane = (r >> 4) & 31;
  const int tile = r >> 9;
  const int KT = Kdim >> 5;
  const int kt = tile % KT;
  const int nt = tile / KT;
  const int krow = kt * 32 + ((lane >> 4) << 3) + (i & 7) + ((i >> 3) << 4);
  const int ncol = nt * 16 + (lane & 15);
  dst[o] = (__bf16)src[(size_t)kk * mat + (size_t)krow * Ndim + ncol];
}

// ---------------------------------------------------------------------------
// Fused edge-message kernel with double-buffered async weight streaming.
// ---------------------------------------------------------------------------
__global__ __launch_bounds__(256) void edge_msg_kernel(
    const float* __restrict__ inputs, const float* __restrict__ rel_type,
    const __bf16* __restrict__ W1p, const float* __restrict__ b1,
    const __bf16* __restrict__ W2p, const float* __restrict__ b2,
    float* __restrict__ agg)
{
  __shared__ __attribute__((aligned(32))) __bf16 wl[2][F2 * NH1 / 2]; // 2x64 KB
  __shared__ __attribute__((aligned(32))) __bf16 premsg[8][16][F2];   // 64 KB
  __shared__ __attribute__((aligned(32))) __bf16 htile[8][16][NH1];   // 64 KB
  __shared__ float rts[8][16][NK];                                    // 2 KB
  __shared__ float parts[8][2][NM];                                   // 16 KB

  const int bid  = blockIdx.x;
  const int b    = bid / (NN / 2);
  const int n0   = (bid % (NN / 2)) * 2;
  const int tid  = threadIdx.x;
  const int wave = tid >> 5;
  const int lane = tid & 31;
  const int h    = lane >> 4;
  const int nc   = lane & 15;
  const int n    = n0 + (wave >> 2);
  const int tile = wave & 3;

  for (int idx = lane; idx < 16 * (F2 / 4); idx += 32) {
    const int t = idx >> 6;
    const int c = (idx & 63) * 4;
    const int jp = tile * 16 + t;
    float4 f = make_float4(0.f, 0.f, 0.f, 0.f);
    if (jp < DEG) {
      const int s = jp + (jp >= n ? 1 : 0);
      const float* sp = (c < NF) ? &inputs[((size_t)b * NN + n) * NF + c]
                                 : &inputs[((size_t)b * NN + s) * NF + (c - NF)];
      f = *(const float4*)sp;
    }
    v4bf pk;
    pk[0] = (__bf16)f.x; pk[1] = (__bf16)f.y; pk[2] = (__bf16)f.z; pk[3] = (__bf16)f.w;
    *(v4bf*)&premsg[wave][t][c] = pk;
  }
  for (int idx = lane; idx < 16 * NK; idx += 32) {
    const int t = idx >> 2, k = idx & 3;
    const int jp = tile * 16 + t;
    rts[wave][t][k] = (jp < DEG)
        ? rel_type[((size_t)b * NE + (size_t)n * DEG + jp) * NK + k] : 0.f;
  }

  float colsum[16];
#pragma unroll
  for (int mt = 0; mt < 16; ++mt) colsum[mt] = 0.f;

  auto layer1_half = [&](const __bf16* wbuf, int ntbase, const float* bias_p) {
    for (int ntl = 0; ntl < 8; ++ntl) {
      const int nt = ntbase + ntl;
      v8f acc = {};
      for (int ks = 0; ks < 8; ++ks) {
        v16bf a  = lda(&premsg[wave][0][0], F2, ks * 32, lane);
        v16bf bb = ldbp(wbuf, 8, ks, ntl, lane);
        acc = __builtin_amdgcn_wmma_f32_16x16x32_bf16(false, a, false, bb,
                                                      (short)0, acc, false, false);
      }
      const float bias = bias_p[nt * 16 + nc];
#pragma unroll
      for (int g = 0; g < 8; ++g) {
        float v = acc[g] + bias;
        v = v > 0.f ? v : 0.f;
        htile[wave][g + 8 * h][nt * 16 + nc] = (__bf16)v;
      }
    }
  };

  auto layer2_half = [&](const __bf16* wbuf, int mtbase, const float* bias_p, int k) {
    for (int mtl = 0; mtl < 8; ++mtl) {
      const int mt = mtbase + mtl;
      v8f acc = {};
      for (int ks = 0; ks < 8; ++ks) {
        v16bf a  = lda(&htile[wave][0][0], NH1, ks * 32, lane);
        v16bf bb = ldbp(wbuf, 8, ks, mtl, lane);
        acc = __builtin_amdgcn_wmma_f32_16x16x32_bf16(false, a, false, bb,
                                                      (short)0, acc, false, false);
      }
      const float bias = bias_p[mt * 16 + nc];
      float part = 0.f;
#pragma unroll
      for (int g = 0; g < 8; ++g) {
        float v = acc[g] + bias;
        v = v > 0.f ? v : 0.f;
        part += v * rts[wave][g + 8 * h][k];
      }
      colsum[mt] += part;
    }
  };

  for (int k = 0; k < NK; ++k) {
    const uint4* W1g = (const uint4*)(W1p + (size_t)k * F2 * NH1);
    const uint4* W2g = (const uint4*)(W2p + (size_t)k * NH1 * NM);
    uint4* bufA = (uint4*)&wl[0][0];
    uint4* bufB = (uint4*)&wl[1][0];
    const float* b1k = b1 + k * NH1;
    const float* b2k = b2 + k * NM;

    stage_half(W1g, bufA, tid);                 // W1.h0
    wait_async_all(); __syncthreads();

    stage_half(W1g + 4096, bufB, tid);          // W1.h1 in flight
    layer1_half(wl[0], 0, b1k);
    wait_async_all(); __syncthreads();

    stage_half(W2g, bufA, tid);                 // W2.h0 in flight
    layer1_half(wl[1], 8, b1k);
    wait_async_all(); __syncthreads();

    stage_half(W2g + 4096, bufB, tid);          // W2.h1 in flight
    layer2_half(wl[0], 0, b2k, k);
    wait_async_all(); __syncthreads();

    if (k + 1 < NK) {                           // warm L2 for next k's W1
      const char* nxt = (const char*)(W1p + (size_t)(k + 1) * F2 * NH1)
                        + (size_t)tid * 512;
      __builtin_prefetch(nxt, 0, 1);
      __builtin_prefetch(nxt + 256, 0, 1);
    }
    layer2_half(wl[1], 8, b2k, k);
    __syncthreads();
  }

#pragma unroll
  for (int mt = 0; mt < 16; ++mt) parts[wave][h][mt * 16 + nc] = colsum[mt];
  __syncthreads();

  for (int idx = tid; idx < 2 * NM; idx += 256) {
    const int which = idx >> 8;
    const int m     = idx & (NM - 1);
    float s = 0.f;
#pragma unroll
    for (int w = 0; w < 4; ++w)
      s += parts[which * 4 + w][0][m] + parts[which * 4 + w][1][m];
    agg[((size_t)b * NN + n0 + which) * NM + m] = s;
  }
}

// ---------------------------------------------------------------------------
// Output MLP kernel
// ---------------------------------------------------------------------------
__global__ __launch_bounds__(32) void out_mlp_kernel(
    const float* __restrict__ inputs, const float* __restrict__ agg,
    const __bf16* __restrict__ Wo1p, const float* __restrict__ bo1,
    const __bf16* __restrict__ Wo2p, const float* __restrict__ bo2,
    const __bf16* __restrict__ Wo3p, const float* __restrict__ bo3,
    float* __restrict__ out)
{
  __shared__ __attribute__((aligned(32))) __bf16 aug[16][NF + NM];
  __shared__ __attribute__((aligned(32))) __bf16 p1[16][NNH];
  __shared__ __attribute__((aligned(32))) __bf16 p2[16][NNH];

  const int row0 = blockIdx.x * 16;
  const int lane = threadIdx.x & 31;
  const int h    = lane >> 4;
  const int nc   = lane & 15;

  for (int idx = lane; idx < 16 * ((NF + NM) / 4); idx += 32) {
    const int t = idx / 96;
    const int c = (idx % 96) * 4;
    const size_t row = (size_t)row0 + t;
    const float* sp = (c < NF) ? &inputs[row * NF + c] : &agg[row * NM + (c - NF)];
    float4 f = *(const float4*)sp;
    v4bf pk;
    pk[0] = (__bf16)f.x; pk[1] = (__bf16)f.y; pk[2] = (__bf16)f.z; pk[3] = (__bf16)f.w;
    *(v4bf*)&aug[t][c] = pk;
  }

  for (int nt = 0; nt < 16; ++nt) {
    v8f acc = {};
    for (int ks = 0; ks < 12; ++ks) {
      v16bf a  = lda(&aug[0][0], NF + NM, ks * 32, lane);
      v16bf bb = ldbp(Wo1p, 12, ks, nt, lane);
      acc = __builtin_amdgcn_wmma_f32_16x16x32_bf16(false, a, false, bb,
                                                    (short)0, acc, false, false);
    }
    const float bias = bo1[nt * 16 + nc];
#pragma unroll
    for (int g = 0; g < 8; ++g) {
      float v = acc[g] + bias;
      v = v > 0.f ? v : 0.f;
      p1[g + 8 * h][nt * 16 + nc] = (__bf16)v;
    }
  }

  for (int nt = 0; nt < 16; ++nt) {
    v8f acc = {};
    for (int ks = 0; ks < 8; ++ks) {
      v16bf a  = lda(&p1[0][0], NNH, ks * 32, lane);
      v16bf bb = ldbp(Wo2p, 8, ks, nt, lane);
      acc = __builtin_amdgcn_wmma_f32_16x16x32_bf16(false, a, false, bb,
                                                    (short)0, acc, false, false);
    }
    const float bias = bo2[nt * 16 + nc];
#pragma unroll
    for (int g = 0; g < 8; ++g) {
      float v = acc[g] + bias;
      v = v > 0.f ? v : 0.f;
      p2[g + 8 * h][nt * 16 + nc] = (__bf16)v;
    }
  }

  for (int nt = 0; nt < 8; ++nt) {
    v8f acc = {};
    for (int ks = 0; ks < 8; ++ks) {
      v16bf a  = lda(&p2[0][0], NNH, ks * 32, lane);
      v16bf bb = ldbp(Wo3p, 8, ks, nt, lane);
      acc = __builtin_amdgcn_wmma_f32_16x16x32_bf16(false, a, false, bb,
                                                    (short)0, acc, false, false);
    }
    const int col = nt * 16 + nc;
    const float bias = bo3[col];
#pragma unroll
    for (int g = 0; g < 8; ++g) {
      const size_t row = (size_t)row0 + g + 8 * h;
      out[row * NF + col] = acc[g] + bias + inputs[row * NF + col];
    }
  }
}

// ---------------------------------------------------------------------------
// host launch
// ---------------------------------------------------------------------------
extern "C" void kernel_launch(void* const* d_in, const int* in_sizes, int n_in,
                              void* d_out, int out_size, void* d_ws, size_t ws_size,
                              hipStream_t stream) {
  (void)in_sizes; (void)n_in; (void)out_size; (void)ws_size;
  const float* inputs   = (const float*)d_in[0];
  const float* rel_type = (const float*)d_in[1];
  const float* W1  = (const float*)d_in[4];
  const float* b1  = (const float*)d_in[5];
  const float* W2  = (const float*)d_in[6];
  const float* b2  = (const float*)d_in[7];
  const float* Wo1 = (const float*)d_in[8];
  const float* bo1 = (const float*)d_in[9];
  const float* Wo2 = (const float*)d_in[10];
  const float* bo2 = (const float*)d_in[11];
  const float* Wo3 = (const float*)d_in[12];
  const float* bo3 = (const float*)d_in[13];

  __bf16* W1p  = (__bf16*)d_ws;
  __bf16* W2p  = W1p  + (size_t)NK * F2 * NH1;
  __bf16* Wo1p = W2p  + (size_t)NK * NH1 * NM;
  __bf16* Wo2p = Wo1p + (size_t)(NF + NM) * NNH;
  __bf16* Wo3p = Wo2p + (size_t)NNH * NNH;
  float*  agg  = (float*)(Wo3p + (size_t)NNH * NF);

  const int n1  = NK * F2 * NH1, n2 = NK * NH1 * NM;
  const int no1 = (NF + NM) * NNH, no2 = NNH * NNH, no3 = NNH * NF;
  pack_b_kernel<<<(n1  + 255) / 256, 256, 0, stream>>>(W1,  W1p,  F2,      NH1, n1);
  pack_b_kernel<<<(n2  + 255) / 256, 256, 0, stream>>>(W2,  W2p,  NH1,     NM,  n2);
  pack_b_kernel<<<(no1 + 255) / 256, 256, 0, stream>>>(Wo1, Wo1p, NF + NM, NNH, no1);
  pack_b_kernel<<<(no2 + 255) / 256, 256, 0, stream>>>(Wo2, Wo2p, NNH,     NNH, no2);
  pack_b_kernel<<<(no3 + 255) / 256, 256, 0, stream>>>(Wo3, Wo3p, NNH,     NF,  no3);

  edge_msg_kernel<<<NB * (NN / 2), 256, 0, stream>>>(inputs, rel_type,
                                                     W1p, b1, W2p, b2, agg);
  out_mlp_kernel<<<(NB * NN) / 16, 32, 0, stream>>>(inputs, agg, Wo1p, bo1,
                                                    Wo2p, bo2, Wo3p, bo3, (float*)d_out);
}
